// LigandLocalEnvEmbedding_12180527251537
// MI455X (gfx1250) — matrix operand order repaired and checked by reading, not compiled, not run on hardware
//
#include <hip/hip_runtime.h>

typedef __attribute__((ext_vector_type(16))) _Float16 v16h;
typedef __attribute__((ext_vector_type(8)))  _Float16 v8h;
typedef __attribute__((ext_vector_type(8)))  float    v8f;

// Problem constants (match reference)
constexpr int   KNN      = 16;
constexpr int   NUM_RBF  = 32;
constexpr float MAX_DIST = 24.0f;
constexpr int   OUT_DIM  = 256;
constexpr int   MM       = 1024;   // atoms per batch
constexpr int   BB       = 64;     // batches
constexpr int   FDIM     = KNN * NUM_RBF;   // 512
constexpr double SPACING_D = 24.0 / 31.0;
constexpr float SPACING = (float)SPACING_D;
constexpr float GAMMA   = (float)(1.0 / (SPACING_D * SPACING_D + 1e-8));

// ---------------------------------------------------------------------------
// Stage 0: convert proj_w [256 x 512] f32 -> f16 (kept L2-resident, 256 KB)
// ---------------------------------------------------------------------------
__global__ __launch_bounds__(256) void cvt_w_f16(const float* __restrict__ w,
                                                 _Float16* __restrict__ wh, int n) {
    int idx = blockIdx.x * 256 + threadIdx.x;
    if (idx < n) wh[idx] = (_Float16)w[idx];
}

// ---------------------------------------------------------------------------
// Stage 1: per-row top-16 nearest squared distances (LDS-tiled scan),
// write sqrt'd + padded distances [B*M, 16] to workspace.
// Register-resident sorted list: insert-at-tail + unrolled min/max bubble,
// all constant indices -> stays in VGPRs (no scratch).
// ---------------------------------------------------------------------------
__global__ __launch_bounds__(256) void topk_kernel(const float* __restrict__ coords,
                                                   const int*  __restrict__ mask,
                                                   float*      __restrict__ topk) {
    __shared__ float4 sj[256];
    const int b     = blockIdx.x >> 2;      // 4 chunks of 256 rows per batch
    const int chunk = blockIdx.x & 3;
    const int t     = threadIdx.x;
    const int i     = chunk * 256 + t;      // row within batch
    const int rowG  = b * MM + i;

    const float xi = coords[(size_t)rowG * 3 + 0];
    const float yi = coords[(size_t)rowG * 3 + 1];
    const float zi = coords[(size_t)rowG * 3 + 2];
    const bool maski = mask[rowG] != 0;
    const float INF = __builtin_inff();

    float best[16];
#pragma unroll
    for (int k = 0; k < 16; ++k) best[k] = INF;

    for (int jt = 0; jt < MM; jt += 256) {
        const int jrow = b * MM + jt + t;
        float4 v;
        v.x = coords[(size_t)jrow * 3 + 0];
        v.y = coords[(size_t)jrow * 3 + 1];
        v.z = coords[(size_t)jrow * 3 + 2];
        v.w = (mask[jrow] != 0) ? 0.0f : INF;   // +inf poisons masked candidates
        __syncthreads();
        sj[t] = v;
        __syncthreads();
        if (maski) {
            for (int jj = 0; jj < 256; ++jj) {
                float4 c = sj[jj];
                float dx = xi - c.x, dy = yi - c.y, dz = zi - c.z;
                float d2 = dx * dx + dy * dy + dz * dz + c.w;
                if (jt + jj == i) d2 = INF;     // exclude self
                if (d2 < best[15]) {
                    best[15] = d2;
#pragma unroll
                    for (int k = 15; k >= 1; --k) {   // one bubble pass restores order
                        float a = best[k - 1], bv = best[k];
                        best[k - 1] = fminf(a, bv);
                        best[k]     = fmaxf(a, bv);
                    }
                }
            }
        }
    }

#pragma unroll
    for (int k = 0; k < 16; ++k) {
        float d = sqrtf(fmaxf(best[k], 1e-12f));
        if (!(d < INF)) d = MAX_DIST;           // pad missing neighbors
        topk[(size_t)rowG * KNN + k] = d;
    }
}

// ---------------------------------------------------------------------------
// Stage 2: WMMA GEMM  out[65536,256] = RBF(topk)[65536,512] @ proj_w^T[512,256]
// Block = 8 waves sharing one 64-column group; B slice (64x512 f16 = 64 KB)
// staged in LDS once. Each wave owns a 16x64 output tile; the inner loop has
// zero global memory ops (LDS B reads + on-the-fly RBF A fragments + WMMA).
// ---------------------------------------------------------------------------
__global__ __launch_bounds__(256) void gemm_kernel(const float*    __restrict__ topk,
                                                   const _Float16* __restrict__ wh,
                                                   const int*      __restrict__ mask,
                                                   float*          __restrict__ out) {
    constexpr int BSTRIDE = FDIM + 8;            // 520 halves: 1040B/col, bank-spread
    __shared__ _Float16 bs[64 * BSTRIDE];        // ~65 KB

    const int lane   = threadIdx.x & 31;
    const int wave   = threadIdx.x >> 5;
    const int colGrp = blockIdx.x & 3;           // 64-col group shared by whole block
    const int rowGrp = blockIdx.x >> 2;
    const int n0     = colGrp * 64;
    const int row0   = (rowGrp * 8 + wave) * 16; // this wave's 16-row tile
    const int laneLo = lane & 15;
    const int hi     = lane >> 4;                // which K-half this lane holds
    const int r      = row0 + laneLo;            // this lane's matrix row

    // --- cooperative stage of B slice: 64 cols x 512 halves (32B chunks) ---
    {
        const int t = threadIdx.x;
#pragma unroll
        for (int it = 0; it < 8; ++it) {
            const int chunk = it * 256 + t;      // 2048 chunks of 16 halves
            const int col   = chunk >> 5;        // 0..63
            const int off   = (chunk & 31) * 16; // halves within column
            v16h v = *(const v16h*)(wh + (size_t)(n0 + col) * FDIM + off);
            v8h lo = __builtin_shufflevector(v, v, 0, 1, 2, 3, 4, 5, 6, 7);
            v8h hh = __builtin_shufflevector(v, v, 8, 9, 10, 11, 12, 13, 14, 15);
            *(v8h*)(bs + col * BSTRIDE + off)     = lo;
            *(v8h*)(bs + col * BSTRIDE + off + 8) = hh;
        }
    }

    // --- hoist this lane's 16 distances (one 64B contiguous run) ---
    float dist[16];
    {
        const float4* dp = (const float4*)(topk + (size_t)r * KNN);
#pragma unroll
        for (int q = 0; q < 4; ++q) {
            float4 v = dp[q];
            dist[4 * q + 0] = v.x; dist[4 * q + 1] = v.y;
            dist[4 * q + 2] = v.z; dist[4 * q + 3] = v.w;
        }
    }
    __syncthreads();

    v8f acc0 = {}, acc1 = {}, acc2 = {}, acc3 = {};

#pragma unroll
    for (int step = 0; step < KNN; ++step) {
        const int kk = step * 32;
        const float d = dist[step];

        // A fragment, contiguous k order (B uses the same):
        //   lanes 0-15 hold K = {0..15}, lanes 16-31 hold K = {16..31}
        v16h a;
#pragma unroll
        for (int p = 0; p < 16; ++p) {
            const int c = p + hi * 16;  // rbf center index
            const float diff = d - (float)c * SPACING;
            a[p] = (_Float16)__expf(-GAMMA * diff * diff);
        }

        // B fragments from LDS: lane = column n, hi selects K half.
        const _Float16* bbase = bs + laneLo * BSTRIDE + kk + hi * 16;
#define LOAD_BFRAG(T) \
        __builtin_shufflevector( \
            *(const v8h*)(bbase + (T) * 16 * BSTRIDE), \
            *(const v8h*)(bbase + (T) * 16 * BSTRIDE + 8), \
            0, 1, 2, 3, 4, 5, 6, 7, 8, 9, 10, 11, 12, 13, 14, 15)
        v16h b0 = LOAD_BFRAG(0);
        v16h b1 = LOAD_BFRAG(1);
        v16h b2 = LOAD_BFRAG(2);
        v16h b3 = LOAD_BFRAG(3);
#undef LOAD_BFRAG

        acc0 = __builtin_amdgcn_wmma_f32_16x16x32_f16(false, a, false, b0, (short)0, acc0, false, false);
        acc1 = __builtin_amdgcn_wmma_f32_16x16x32_f16(false, a, false, b1, (short)0, acc1, false, false);
        acc2 = __builtin_amdgcn_wmma_f32_16x16x32_f16(false, a, false, b2, (short)0, acc2, false, false);
        acc3 = __builtin_amdgcn_wmma_f32_16x16x32_f16(false, a, false, b3, (short)0, acc3, false, false);
    }

    // C/D layout: VGPR g, lanes 0-15 -> row g, lanes 16-31 -> row g+8; col = laneLo
#pragma unroll
    for (int g = 0; g < 8; ++g) {
        const int rm = row0 + g + hi * 8;
        const float mk = (mask[rm] != 0) ? 1.0f : 0.0f;
        float* orow = out + (size_t)rm * OUT_DIM + n0 + laneLo;
        orow[0 * 16] = acc0[g] * mk;
        orow[1 * 16] = acc1[g] * mk;
        orow[2 * 16] = acc2[g] * mk;
        orow[3 * 16] = acc3[g] * mk;
    }
}

// ---------------------------------------------------------------------------
extern "C" void kernel_launch(void* const* d_in, const int* in_sizes, int n_in,
                              void* d_out, int out_size, void* d_ws, size_t ws_size,
                              hipStream_t stream) {
    const float* coords = (const float*)d_in[0];   // [64,1024,3] f32
    const int*   maskp  = (const int*)d_in[1];     // [64,1024]
    const float* proj_w = (const float*)d_in[2];   // [256,512] f32
    float*       out    = (float*)d_out;           // [64,1024,256] f32

    // workspace: [0, 4MB) top-16 distances f32; [4MB, +256KB) proj_w f16
    float*    topk = (float*)d_ws;
    _Float16* wh   = (_Float16*)((char*)d_ws + (size_t)BB * MM * KNN * sizeof(float));

    cvt_w_f16<<<(OUT_DIM * FDIM + 255) / 256, 256, 0, stream>>>(proj_w, wh, OUT_DIM * FDIM);
    topk_kernel<<<BB * (MM / 256), 256, 0, stream>>>(coords, maskp, topk);
    gemm_kernel<<<(BB * MM / 16) * 4 / 8, 256, 0, stream>>>(topk, wh, maskp, out);
}
